// HolographicMultiheadAttention_30253749633615
// MI455X (gfx1250) — matrix operand order repaired and checked by reading, not compiled, not run on hardware
//
#include <hip/hip_runtime.h>
#include <hip/hip_bf16.h>
#include <math.h>

#define B_   2
#define T_   1024
#define D_   1024
#define H_   16
#define DK_  64
#define EPS_ 1e-8f

typedef __bf16 bf16_t;
typedef __attribute__((ext_vector_type(8)))  __bf16 bf16x8;
typedef __attribute__((ext_vector_type(16))) __bf16 bf16x16;
typedef __attribute__((ext_vector_type(8)))  float  f32x8;

union BFrag {
    bf16x16  v;
    bf16x8   h[2];
    unsigned u[8];
};

__device__ __forceinline__ f32x8 zero8() {
    f32x8 z = {0.f, 0.f, 0.f, 0.f, 0.f, 0.f, 0.f, 0.f};
    return z;
}

// Wave-local LDS ordering: all prior DS ops complete; compiler may not reorder
// LDS accesses across this point. (Per-wave LDS buffers -> no block barrier.)
__device__ __forceinline__ void lds_fence_wave() {
    asm volatile("s_wait_dscnt 0x0" ::: "memory");
}

// A-matrix 16x32 bf16 fragment (M=i rows, K contiguous in memory, row stride ld).
// Lane L<16: M=L, K chunks {0..7, 16..23}; Lane L>=16: M=L-16, K chunks {8..15, 24..31}.
__device__ __forceinline__ BFrag load_a_frag(const bf16_t* p, int ld, int lane) {
    BFrag f;
    int m  = lane & 15;
    int c0 = (lane & 16) ? 8 : 0;
    const bf16_t* row = p + (long)m * ld;
    f.h[0] = *(const bf16x8*)(row + c0);
    f.h[1] = *(const bf16x8*)(row + c0 + 16);
    return f;
}

// B-matrix 32x16 bf16 fragment. B(k,n) = p[n*ld + k] (columns of B are contiguous rows
// of the source). Lane L<16: N=L, K chunks {0..7, 16..23}; Lane L>=16: N=L-16, K chunks {8..15, 24..31}.
__device__ __forceinline__ BFrag load_b_frag(const bf16_t* p, int ld, int lane) {
    BFrag f;
    int n   = lane & 15;
    int klo = (lane & 16) ? 8 : 0;
    const bf16_t* col = p + (long)n * ld + klo;
    f.h[0] = *(const bf16x8*)(col);
    f.h[1] = *(const bf16x8*)(col + 16);
    return f;
}

__device__ __forceinline__ BFrag bneg(BFrag a) {
    BFrag r;
#pragma unroll
    for (int i = 0; i < 8; ++i) r.u[i] = a.u[i] ^ 0x80008000u;
    return r;
}

__device__ __forceinline__ f32x8 wmma_bf16(const BFrag& a, const BFrag& b, f32x8 c) {
    return __builtin_amdgcn_wmma_f32_16x16x32_bf16(false, a.v, false, b.v,
                                                   (short)0, c, false, false);
}

__device__ __forceinline__ float wrap_angle(float a) {
    const float PI_F   = 3.14159265358979323846f;
    const float TWO_PI = 6.28318530717958647692f;
    const float INV2PI = 0.15915494309189533577f;
    float t = a + PI_F;
    t = t - TWO_PI * floorf(t * INV2PI);
    return t - PI_F;
}

// ---------------------------------------------------------------- conversion
__global__ void k_f2bf(const float* __restrict__ s, bf16_t* __restrict__ d, long n) {
    long i      = (long)blockIdx.x * blockDim.x + threadIdx.x;
    long stride = (long)gridDim.x * blockDim.x;
    for (; i < n; i += stride) d[i] = (bf16_t)s[i];
}

// -------------------------------------------------------------- head norms
// One thread per (b,h,t) row: sqrt(sum |Q|^2) over DK, vectorized bf16x8 loads.
__global__ void k_norms(const bf16_t* __restrict__ Xr, const bf16_t* __restrict__ Xi,
                        float* __restrict__ nrm, long nrows) {
    long row = (long)blockIdx.x * blockDim.x + threadIdx.x;
    if (row >= nrows) return;
    const bf16_t* pr = Xr + row * DK_;
    const bf16_t* pi = Xi + row * DK_;
    float s = 0.f;
#pragma unroll
    for (int c = 0; c < DK_ / 8; ++c) {
        bf16x8 a = *(const bf16x8*)(pr + c * 8);
        bf16x8 b = *(const bf16x8*)(pi + c * 8);
#pragma unroll
        for (int e = 0; e < 8; ++e) {
            float x = (float)a[e], y = (float)b[e];
            s += x * x + y * y;
        }
    }
    nrm[row] = sqrtf(s);
}

// -------------------------------------------- complex linear -> head layout
// y = x @ (Wr + i Wi)^T + (br + i bi); x = (ar + i ai), all [2048 x 1024].
// mode 0: store [B,H,T,DK]; mode 1: store V transposed [B,H,DK,T].
__global__ void __launch_bounds__(128)
k_clin(const bf16_t* __restrict__ ar_, const bf16_t* __restrict__ ai_,
       const bf16_t* __restrict__ wr_, const bf16_t* __restrict__ wi_,
       const float* __restrict__ br_, const float* __restrict__ bi_,
       bf16_t* __restrict__ outR, bf16_t* __restrict__ outI, int mode) {
    const int lane = threadIdx.x & 31;
    const int wave = threadIdx.x >> 5;
    const int tile = blockIdx.x * 4 + wave;   // 8192 tiles = 128 x 64
    const int m0   = (tile >> 6) * 16;
    const int n0   = (tile & 63) * 16;

    f32x8 accR = zero8();
    f32x8 accI = zero8();

    for (int k = 0; k < D_; k += 32) {
        // prefetch streamed A rows two k-steps ahead into near caches
        __builtin_prefetch(ar_ + (long)m0 * D_ + k + 64, 0, 3);
        __builtin_prefetch(ai_ + (long)m0 * D_ + k + 64, 0, 3);
        BFrag Ar  = load_a_frag(ar_ + (long)m0 * D_ + k, D_, lane);
        BFrag Ai  = load_a_frag(ai_ + (long)m0 * D_ + k, D_, lane);
        BFrag Ain = bneg(Ai);
        BFrag Br  = load_b_frag(wr_ + (long)n0 * D_ + k, D_, lane);
        BFrag Bi  = load_b_frag(wi_ + (long)n0 * D_ + k, D_, lane);
        accR = wmma_bf16(Ar,  Br, accR);   // + xr*Wr
        accR = wmma_bf16(Ain, Bi, accR);   // - xi*Wi
        accI = wmma_bf16(Ar,  Bi, accI);   // + xr*Wi
        accI = wmma_bf16(Ai,  Br, accI);   // + xi*Wr
    }

    const int   nc    = n0 + (lane & 15);
    const float bR    = br_[nc];
    const float bI    = bi_[nc];
    const int   rbase = (lane & 16) ? 8 : 0;
    const int   h     = nc >> 6;
    const int   dk    = nc & 63;
#pragma unroll
    for (int r = 0; r < 8; ++r) {
        int  m = m0 + r + rbase;
        int  b = m >> 10;
        int  t = m & 1023;
        long idx = (mode == 0)
                       ? ((long)((b * H_ + h) * T_ + t) * DK_ + dk)
                       : ((long)((b * H_ + h) * DK_ + dk) * T_ + t);
        outR[idx] = (bf16_t)(accR[r] + bR);
        outI[idx] = (bf16_t)(accI[r] + bI);
    }
}

// ------------------------------------------------- fused holographic attention
__global__ void __launch_bounds__(128)
k_attn(const bf16_t* __restrict__ Qr, const bf16_t* __restrict__ Qi,
       const bf16_t* __restrict__ Kr, const bf16_t* __restrict__ Ki,
       const bf16_t* __restrict__ Vtr, const bf16_t* __restrict__ Vti,
       const float* __restrict__ qn_g, const float* __restrict__ kn_g,
       const float* __restrict__ alpha_param, const unsigned char* __restrict__ mask,
       bf16_t* __restrict__ aor, bf16_t* __restrict__ aoi) {
    __shared__ bf16_t lds_pc[4][16 * 32];
    __shared__ bf16_t lds_ps[4][16 * 32];

    const int  lane = threadIdx.x & 31;
    const int  wave = threadIdx.x >> 5;
    const int  g    = blockIdx.x * 4 + wave;  // 2048 i-tiles total
    const int  bh   = g >> 6;                 // 0..31
    const int  t0   = (g & 63) * 16;
    const int  b    = bh >> 4;
    const int  h    = bh & 15;
    const long hoff = (long)bh * T_ * DK_;

    const float alpha = log1pf(__expf(alpha_param[h])) + 1e-6f;

    // Resident Q fragments (DK=64 -> two K=32 steps), plus negated real parts.
    BFrag qr[2], qi[2], qrn[2];
#pragma unroll
    for (int ks = 0; ks < 2; ++ks) {
        qr[ks]  = load_a_frag(Qr + hoff + (long)t0 * DK_ + ks * 32, DK_, lane);
        qi[ks]  = load_a_frag(Qi + hoff + (long)t0 * DK_ + ks * 32, DK_, lane);
        qrn[ks] = bneg(qr[ks]);
    }

    // Per-row q norms from the precomputed table (rows match C-layout mapping).
    const int ib = (lane & 16) ? 8 : 0;
    const int jA = (lane & 15);
    float qnarr[8];
#pragma unroll
    for (int r = 0; r < 8; ++r) qnarr[r] = qn_g[bh * T_ + t0 + r + ib];

    float m_run[8], l_run[8];
#pragma unroll
    for (int r = 0; r < 8; ++r) { m_run[r] = -1e30f; l_run[r] = 0.f; }
    f32x8 accR[4], accI[4];
#pragma unroll
    for (int nt = 0; nt < 4; ++nt) { accR[nt] = zero8(); accI[nt] = zero8(); }

    for (int j0 = 0; j0 < T_; j0 += 32) {
        // Per-column k-norm + padding mask, directly at this lane's j columns.
        float kn_h[2], mk_h[2];
#pragma unroll
        for (int jh = 0; jh < 2; ++jh) {
            int j = j0 + jA + 16 * jh;
            kn_h[jh] = kn_g[bh * T_ + j];
            mk_h[jh] = mask[b * T_ + j] ? 1.f : 0.f;
        }

        // Complex scores S = Q . conj(K) for two 16-wide j halves.
        float wv[2][8], dp[2][8];
#pragma unroll
        for (int jh = 0; jh < 2; ++jh) {
            const bf16_t* kbR = Kr + hoff + (long)(j0 + jh * 16) * DK_;
            const bf16_t* kbI = Ki + hoff + (long)(j0 + jh * 16) * DK_;
            BFrag Kr0 = load_b_frag(kbR, DK_, lane);
            BFrag Kr1 = load_b_frag(kbR + 32, DK_, lane);
            BFrag Ki0 = load_b_frag(kbI, DK_, lane);
            BFrag Ki1 = load_b_frag(kbI + 32, DK_, lane);
            f32x8 sr = zero8(), si = zero8();
            sr = wmma_bf16(qr[0], Kr0, sr);   // Qr.Kr
            sr = wmma_bf16(qi[0], Ki0, sr);   // +Qi.Ki
            sr = wmma_bf16(qr[1], Kr1, sr);
            sr = wmma_bf16(qi[1], Ki1, sr);
            si = wmma_bf16(qi[0], Kr0, si);   // Qi.Kr
            si = wmma_bf16(qrn[0], Ki0, si);  // -Qr.Ki
            si = wmma_bf16(qi[1], Kr1, si);
            si = wmma_bf16(qrn[1], Ki1, si);
#pragma unroll
            for (int r = 0; r < 8; ++r) {
                float srv = sr[r], siv = si[r];
                float ph  = wrap_angle(atan2f(siv + EPS_, srv + EPS_));
                float sim = srv / (qnarr[r] * kn_h[jh] + EPS_);
                float w   = sim * 0.125f * __expf(-alpha * fabsf(ph));
                w         = (mk_h[jh] != 0.f) ? -1e9f : w;
                wv[jh][r] = w;
                dp[jh][r] = ph;
            }
        }

        // Online softmax + phasor-weighted P tiles into LDS (C-layout -> A-layout).
        bf16_t* pc = &lds_pc[wave][0];
        bf16_t* ps = &lds_ps[wave][0];
#pragma unroll
        for (int r = 0; r < 8; ++r) {
            float mt = fmaxf(wv[0][r], wv[1][r]);
            mt = fmaxf(mt, __shfl_xor(mt, 1));
            mt = fmaxf(mt, __shfl_xor(mt, 2));
            mt = fmaxf(mt, __shfl_xor(mt, 4));
            mt = fmaxf(mt, __shfl_xor(mt, 8));
            float mn = fmaxf(m_run[r], mt);
            float sc = __expf(m_run[r] - mn);
            m_run[r] = mn;
            float p0 = __expf(wv[0][r] - mn);
            float p1 = __expf(wv[1][r] - mn);
            float rs = p0 + p1;
            rs += __shfl_xor(rs, 1);
            rs += __shfl_xor(rs, 2);
            rs += __shfl_xor(rs, 4);
            rs += __shfl_xor(rs, 8);
            l_run[r] = l_run[r] * sc + rs;
#pragma unroll
            for (int nt = 0; nt < 4; ++nt) { accR[nt][r] *= sc; accI[nt][r] *= sc; }
            float s0, c0, s1, c1;
            __sincosf(dp[0][r], &s0, &c0);
            __sincosf(dp[1][r], &s1, &c1);
            int i_loc = r + ib;
            pc[i_loc * 32 + jA]      = (bf16_t)(p0 * c0);
            ps[i_loc * 32 + jA]      = (bf16_t)(p0 * s0);
            pc[i_loc * 32 + jA + 16] = (bf16_t)(p1 * c1);
            ps[i_loc * 32 + jA + 16] = (bf16_t)(p1 * s1);
        }
        // Wave-local transpose buffers: DS ops are in-order per wave, so a
        // dscnt wait (not a block barrier) orders stores -> loads.
        lds_fence_wave();

        BFrag Pc  = load_a_frag(pc, 32, lane);
        BFrag Ps  = load_a_frag(ps, 32, lane);
        BFrag Psn = bneg(Ps);
#pragma unroll
        for (int nt = 0; nt < 4; ++nt) {
            const bf16_t* vbR = Vtr + ((long)bh * DK_ + nt * 16) * T_ + j0;
            const bf16_t* vbI = Vti + ((long)bh * DK_ + nt * 16) * T_ + j0;
            BFrag Vr = load_b_frag(vbR, T_, lane);
            BFrag Vi = load_b_frag(vbI, T_, lane);
            accR[nt] = wmma_bf16(Pc,  Vr, accR[nt]);   // + Pcos.Vr
            accR[nt] = wmma_bf16(Psn, Vi, accR[nt]);   // - Psin.Vi
            accI[nt] = wmma_bf16(Pc,  Vi, accI[nt]);   // + Pcos.Vi
            accI[nt] = wmma_bf16(Ps,  Vr, accI[nt]);   // + Psin.Vr
        }
        // Order this tile's LDS reads before next tile's overwrites.
        lds_fence_wave();
    }

    // Normalize and store merged-head layout [B,T,D] bf16.
#pragma unroll
    for (int nt = 0; nt < 4; ++nt) {
        int dcol = h * DK_ + nt * 16 + jA;
#pragma unroll
        for (int r = 0; r < 8; ++r) {
            int   t   = t0 + r + ib;
            float inv = 1.f / l_run[r];
            long  idx = ((long)(b * T_ + t)) * D_ + dcol;
            aor[idx] = (bf16_t)(accR[nt][r] * inv);
            aoi[idx] = (bf16_t)(accI[nt][r] * inv);
        }
    }
}

// --------------------------------------------------- output complex linear
__global__ void __launch_bounds__(128)
k_proj_out(const bf16_t* __restrict__ ar_, const bf16_t* __restrict__ ai_,
           const bf16_t* __restrict__ wr_, const bf16_t* __restrict__ wi_,
           const float* __restrict__ br_, const float* __restrict__ bi_,
           float* __restrict__ out) {
    const int lane = threadIdx.x & 31;
    const int wave = threadIdx.x >> 5;
    const int tile = blockIdx.x * 4 + wave;
    const int m0   = (tile >> 6) * 16;
    const int n0   = (tile & 63) * 16;

    f32x8 accR = zero8();
    f32x8 accI = zero8();

    for (int k = 0; k < D_; k += 32) {
        __builtin_prefetch(ar_ + (long)m0 * D_ + k + 64, 0, 3);
        __builtin_prefetch(ai_ + (long)m0 * D_ + k + 64, 0, 3);
        BFrag Ar  = load_a_frag(ar_ + (long)m0 * D_ + k, D_, lane);
        BFrag Ai  = load_a_frag(ai_ + (long)m0 * D_ + k, D_, lane);
        BFrag Ain = bneg(Ai);
        BFrag Br  = load_b_frag(wr_ + (long)n0 * D_ + k, D_, lane);
        BFrag Bi  = load_b_frag(wi_ + (long)n0 * D_ + k, D_, lane);
        accR = wmma_bf16(Ar,  Br, accR);
        accR = wmma_bf16(Ain, Bi, accR);
        accI = wmma_bf16(Ar,  Bi, accI);
        accI = wmma_bf16(Ai,  Br, accI);
    }

    const int   nc    = n0 + (lane & 15);
    const float bR    = br_[nc];
    const float bI    = bi_[nc];
    const int   rbase = (lane & 16) ? 8 : 0;
#pragma unroll
    for (int r = 0; r < 8; ++r) {
        int  m   = m0 + r + rbase;
        long idx = ((long)m * D_ + nc) * 2;
        out[idx]     = accR[r] + bR;
        out[idx + 1] = accI[r] + bI;
    }
}

// ------------------------------------------------------------------- launch
extern "C" void kernel_launch(void* const* d_in, const int* in_sizes, int n_in,
                              void* d_out, int out_size, void* d_ws, size_t ws_size,
                              hipStream_t stream) {
    const float* x_real = (const float*)d_in[0];
    const float* x_imag = (const float*)d_in[1];
    const float* wq_r   = (const float*)d_in[2];
    const float* wq_i   = (const float*)d_in[3];
    const float* bq_r   = (const float*)d_in[4];
    const float* bq_i   = (const float*)d_in[5];
    const float* wk_r   = (const float*)d_in[6];
    const float* wk_i   = (const float*)d_in[7];
    const float* bk_r   = (const float*)d_in[8];
    const float* bk_i   = (const float*)d_in[9];
    const float* wv_r   = (const float*)d_in[10];
    const float* wv_i   = (const float*)d_in[11];
    const float* bv_r   = (const float*)d_in[12];
    const float* bv_i   = (const float*)d_in[13];
    const float* wo_r   = (const float*)d_in[14];
    const float* wo_i   = (const float*)d_in[15];
    const float* bo_r   = (const float*)d_in[16];
    const float* bo_i   = (const float*)d_in[17];
    const float* alphap = (const float*)d_in[18];
    const unsigned char* mask = (const unsigned char*)d_in[19];

    const long NX = (long)B_ * T_ * D_;   // 2,097,152
    const long NW = (long)D_ * D_;        // 1,048,576
    const long NR = (long)B_ * H_ * T_;   // 32,768 head-rows

    bf16_t* w = (bf16_t*)d_ws;
    bf16_t* xr_bf = w;  w += NX;
    bf16_t* xi_bf = w;  w += NX;
    bf16_t* wqr_bf = w; w += NW;
    bf16_t* wqi_bf = w; w += NW;
    bf16_t* wkr_bf = w; w += NW;
    bf16_t* wki_bf = w; w += NW;
    bf16_t* wvr_bf = w; w += NW;
    bf16_t* wvi_bf = w; w += NW;
    bf16_t* wor_bf = w; w += NW;
    bf16_t* woi_bf = w; w += NW;
    bf16_t* Qr  = w; w += NX;
    bf16_t* Qi  = w; w += NX;
    bf16_t* Kr  = w; w += NX;
    bf16_t* Ki  = w; w += NX;
    bf16_t* Vtr = w; w += NX;
    bf16_t* Vti = w; w += NX;
    bf16_t* aor = w; w += NX;
    bf16_t* aoi = w; w += NX;
    float*  qn_g = (float*)w;  w += 2 * NR;  // NR floats
    float*  kn_g = (float*)w;  w += 2 * NR;  // NR floats

    // fp32 -> bf16 conversions
    k_f2bf<<<1024, 256, 0, stream>>>(x_real, xr_bf, NX);
    k_f2bf<<<1024, 256, 0, stream>>>(x_imag, xi_bf, NX);
    k_f2bf<<<1024, 256, 0, stream>>>(wq_r, wqr_bf, NW);
    k_f2bf<<<1024, 256, 0, stream>>>(wq_i, wqi_bf, NW);
    k_f2bf<<<1024, 256, 0, stream>>>(wk_r, wkr_bf, NW);
    k_f2bf<<<1024, 256, 0, stream>>>(wk_i, wki_bf, NW);
    k_f2bf<<<1024, 256, 0, stream>>>(wv_r, wvr_bf, NW);
    k_f2bf<<<1024, 256, 0, stream>>>(wv_i, wvi_bf, NW);
    k_f2bf<<<1024, 256, 0, stream>>>(wo_r, wor_bf, NW);
    k_f2bf<<<1024, 256, 0, stream>>>(wo_i, woi_bf, NW);

    // Q, K: [B,H,T,DK];  V: transposed [B,H,DK,T]
    k_clin<<<2048, 128, 0, stream>>>(xr_bf, xi_bf, wqr_bf, wqi_bf, bq_r, bq_i, Qr, Qi, 0);
    k_clin<<<2048, 128, 0, stream>>>(xr_bf, xi_bf, wkr_bf, wki_bf, bk_r, bk_i, Kr, Ki, 0);
    k_clin<<<2048, 128, 0, stream>>>(xr_bf, xi_bf, wvr_bf, wvi_bf, bv_r, bv_i, Vtr, Vti, 1);

    // Precomputed row norms for Q and K
    k_norms<<<(int)((NR + 255) / 256), 256, 0, stream>>>(Qr, Qi, qn_g, NR);
    k_norms<<<(int)((NR + 255) / 256), 256, 0, stream>>>(Kr, Ki, kn_g, NR);

    // Fused attention: 2048 i-tiles, 4 waves/block
    k_attn<<<512, 128, 0, stream>>>(Qr, Qi, Kr, Ki, Vtr, Vti, qn_g, kn_g,
                                    alphap, mask, aor, aoi);

    // Output projection -> stacked [B,T,D,2] fp32
    k_proj_out<<<2048, 128, 0, stream>>>(aor, aoi, wor_bf, woi_bf, bo_r, bo_i,
                                         (float*)d_out);
}
